// PolicyNetwork_34857954574393
// MI455X (gfx1250) — matrix operand order repaired and checked by reading, not compiled, not run on hardware
//
#include <hip/hip_runtime.h>
#include <hip/hip_bf16.h>
#include <math.h>

// ---- sizes ----
// B=256, A=3, N=1000, D=128, H=8 (dk=16), FF=512
#define cB   256
#define cN   1000
#define cD   128
#define cH   8
#define cFF  512
#define cA   3

typedef __attribute__((ext_vector_type(16))) __bf16 v16bf;
typedef __attribute__((ext_vector_type(2)))  __bf16 bf16x2;
typedef __attribute__((ext_vector_type(8)))  float  v8f;

__device__ __forceinline__ v8f wmma_bf16(v16bf a, v16bf b, v8f c) {
  // D = A(16x32) * B(32x16) + C, f32 accum
  return __builtin_amdgcn_wmma_f32_16x16x32_bf16(false, a, false, b, (short)0, c,
                                                 false, false);
}

__device__ __forceinline__ v8f zero8() {
  v8f z;
#pragma unroll
  for (int i = 0; i < 8; ++i) z[i] = 0.0f;
  return z;
}
__device__ __forceinline__ v16bf zero16bf() {
  v16bf z;
#pragma unroll
  for (int i = 0; i < 16; ++i) z[i] = (__bf16)0.0f;
  return z;
}

// ---- A fragment (16x32 bf16) from bf16 row-major A[m][k] ----
// lane L: row m = m0 + (L&15); VGPR v holds K pair k = k0 + (L>>4)*16 + v*2
__device__ __forceinline__ v16bf load_a_bf16(const __bf16* __restrict__ A, int lda,
                                             int m0, int k0, int mmax, int lane) {
  v16bf a;
  int m  = m0 + (lane & 15);
  int hi = lane >> 4;
  bool ok = (m < mmax);
  const __bf16* row = A + (size_t)m * lda;
#pragma unroll
  for (int v = 0; v < 8; ++v) {
    int k = k0 + (hi << 4) + (v << 1);
    if (ok) {
      bf16x2 t = *(const bf16x2*)(row + k);
      a[2 * v] = t[0]; a[2 * v + 1] = t[1];
    } else {
      a[2 * v] = (__bf16)0.f; a[2 * v + 1] = (__bf16)0.f;
    }
  }
  return a;
}

// ---- B fragment (32x16 bf16) for C = A * W^T, W row-major bf16 [ncols][K] ----
// B[k][n] = W[n][k]; lane L: col n = n0 + (L&15); k = k0 + (L>>4)*16 + 2v
// 16 contiguous bf16 per lane -> two 16B vector loads, zero VALU.
__device__ __forceinline__ v16bf load_b_bf16(const __bf16* __restrict__ W, int ldw,
                                             int n0, int k0, int lane) {
  v16bf b;
  int n  = n0 + (lane & 15);
  int kb = k0 + ((lane >> 4) << 4);
  const __bf16* p = W + (size_t)n * ldw + kb;
#pragma unroll
  for (int v = 0; v < 8; ++v) {
    bf16x2 t = *(const bf16x2*)(p + 2 * v);
    b[2 * v] = t[0]; b[2 * v + 1] = t[1];
  }
  return b;
}

// wave computes a 16x128 strip: acc[nt] += A(m0..,:K) * W(n0g+16nt..,:K)^T
__device__ __forceinline__ void mm_strip(const __bf16* __restrict__ A, int lda,
                                         int m0, int mmax,
                                         const __bf16* __restrict__ W, int ldw,
                                         int n0g, int K, int lane, v8f acc[8]) {
  for (int kk = 0; kk < K; kk += 32) {
    v16bf a = load_a_bf16(A, lda, m0, kk, mmax, lane);
#pragma unroll
    for (int nt = 0; nt < 8; ++nt) {
      v16bf b = load_b_bf16(W, ldw, n0g + nt * 16, kk, lane);
      acc[nt] = wmma_bf16(a, b, acc[nt]);
    }
  }
}

// ================= stage 0: weights f32 -> bf16 (run once per call) ===========
__global__ void k_cvtw(const float* __restrict__ src, __bf16* __restrict__ dst, int n) {
  int i = blockIdx.x * 256 + threadIdx.x;
  if (i < n) dst[i] = (__bf16)src[i];
}

// ================= stage 1: features + W_na embed -> h f32 + h bf16 ===========
__global__ void k_prep(const float* __restrict__ loc, const float* __restrict__ speed,
                       const int* __restrict__ next_agent, const int* __restrict__ position,
                       const float* __restrict__ tta, const float* __restrict__ loadv,
                       const float* __restrict__ demand, const float* __restrict__ W_na,
                       const float* __restrict__ b_na, float* __restrict__ h,
                       __bf16* __restrict__ h16) {
  int b = blockIdx.y;
  int d = threadIdx.x;                 // 0..127
  int na = next_agent[b];
  int perm[3]; perm[0] = na;
  { int t = 1; for (int a = 0; a < cA; ++a) if (a != na) perm[t++] = a; }
  float ld[3], sp[3], tt[3], ax[3], ay[3];
#pragma unroll
  for (int i = 0; i < 3; ++i) {
    int ai = perm[i];
    ld[i] = loadv[b * cA + ai];
    sp[i] = speed[b * cA + ai];
    tt[i] = tta[b * cA + ai];
    int pos = position[b * cA + ai];
    ax[i] = loc[(size_t)b * cN * 2 + (size_t)pos * 2];
    ay[i] = loc[(size_t)b * cN * 2 + (size_t)pos * 2 + 1];
  }
  float w[6];
#pragma unroll
  for (int i = 0; i < 6; ++i) w[i] = W_na[d * 6 + i];
  float bias = b_na[d];
  int nbase = blockIdx.x * 8;
  for (int nn = 0; nn < 8; ++nn) {
    int n = nbase + nn;
    if (n >= cN) break;
    float dem = demand[b * cN + n];
    float lx = loc[(size_t)b * cN * 2 + (size_t)n * 2];
    float ly = loc[(size_t)b * cN * 2 + (size_t)n * 2 + 1];
    float acc = bias;
#pragma unroll
    for (int i = 0; i < 3; ++i) {
      float dr = dem / ld[i];
      if (dr > 1.0f) dr = -1.0f;
      if (!(dr == dr)) dr = 0.0f;      // nan_to_num
      float dx = ax[i] - lx, dy = ay[i] - ly;
      float dist = sqrtf(dx * dx + dy * dy);
      float tn = (tt[i] == -1.0f) ? -1.0f : dist / sp[i] + tt[i];
      acc += w[i] * dr + w[3 + i] * tn;
    }
    size_t idx = (size_t)b * cN * cD + (size_t)n * cD + d;
    h[idx]   = acc;
    h16[idx] = (__bf16)acc;
  }
}

// ================= stage 2: QKV projections (WMMA) ============================
__global__ void k_qkv(const __bf16* __restrict__ h16,
                      const __bf16* __restrict__ Wq16, const float* __restrict__ bq,
                      const __bf16* __restrict__ Wk16, const float* __restrict__ bk,
                      const __bf16* __restrict__ Wv16, const float* __restrict__ bv,
                      __bf16* __restrict__ qo, __bf16* __restrict__ ko,
                      __bf16* __restrict__ vo) {
  const __bf16* W; const float* bias; __bf16* out;
  if (blockIdx.z == 0)      { W = Wq16; bias = bq; out = qo; }
  else if (blockIdx.z == 1) { W = Wk16; bias = bk; out = ko; }
  else                      { W = Wv16; bias = bv; out = vo; }
  int b = blockIdx.y, lane = threadIdx.x & 31, wid = threadIdx.x >> 5;
  int m0 = blockIdx.x * 128 + wid * 16;
  const __bf16* A = h16 + (size_t)b * cN * cD;
  v8f acc[8];
#pragma unroll
  for (int i = 0; i < 8; ++i) acc[i] = zero8();
  mm_strip(A, cD, m0, cN, W, cD, 0, cD, lane, acc);
  int col = lane & 15, hi = lane >> 4, mb = m0 + hi * 8;
  __bf16* ob = out + (size_t)b * cN * cD;
#pragma unroll
  for (int nt = 0; nt < 8; ++nt) {
    int n = nt * 16 + col;
    float bv_ = bias[n];
#pragma unroll
    for (int r = 0; r < 8; ++r) {
      int m = mb + r;
      if (m < cN) ob[(size_t)m * cD + n] = (__bf16)(acc[nt][r] + bv_);
    }
  }
}

// ================= stage 3: flash attention (WMMA, dk=16 padded to K=32) ======
__global__ void k_flash(const __bf16* __restrict__ Q, const __bf16* __restrict__ K,
                        const __bf16* __restrict__ V, __bf16* __restrict__ O) {
  __shared__ __bf16 psh[4][256];       // one 16x16 P tile per wave
  int lane = threadIdx.x & 31, wid = threadIdx.x >> 5;
  int qt = blockIdx.x * 4 + wid;
  if (qt >= 63) return;                // ceil(1000/16) = 63 tiles
  int head = blockIdx.y, b = blockIdx.z;
  int m0 = qt * 16;
  const __bf16* Qb = Q + (size_t)b * cN * cD;
  const __bf16* Kb = K + (size_t)b * cN * cD;
  const __bf16* Vb = V + (size_t)b * cN * cD;
  int col = lane & 15, hi = lane >> 4;

  // Q fragment (rows m, dims 0..15 in both lane halves; kf zeroes k=16..31)
  v16bf qa = zero16bf();
  {
    int m = m0 + col;
    if (m < cN) {
      const __bf16* qp = Qb + (size_t)m * cD + head * 16;
#pragma unroll
      for (int v = 0; v < 8; ++v) {
        int k = v * 2;
        bf16x2 t = *(const bf16x2*)(qp + k);
        qa[2 * v] = t[0]; qa[2 * v + 1] = t[1];
      }
    }
  }
  v8f oacc = zero8();
  float mrun[8], lrun[8];
#pragma unroll
  for (int r = 0; r < 8; ++r) { mrun[r] = -1e30f; lrun[r] = 0.f; }

  for (int jt = 0; jt < 63; ++jt) {
    int j0 = jt * 16;
    int j = j0 + col;                  // key column handled by this lane
    // K^T fragment: B[k][j] = K[j][k]; only lanes hi==0 carry k=0..15
    v16bf kf = zero16bf();
    if (hi == 0 && j < cN) {
      const __bf16* kp = Kb + (size_t)j * cD + head * 16;
#pragma unroll
      for (int v = 0; v < 8; ++v) {
        bf16x2 t = *(const bf16x2*)(kp + 2 * v);
        kf[2 * v] = t[0]; kf[2 * v + 1] = t[1];
      }
    }
    v8f s = wmma_bf16(qa, kf, zero8());
    float sv[8];
    bool valid = (j < cN);
#pragma unroll
    for (int r = 0; r < 8; ++r) sv[r] = valid ? s[r] * 0.25f : -1e30f;
    // row max across the 16 lanes of each half-wave
    float rmax[8];
#pragma unroll
    for (int r = 0; r < 8; ++r) rmax[r] = sv[r];
    for (int off = 1; off < 16; off <<= 1) {
#pragma unroll
      for (int r = 0; r < 8; ++r) rmax[r] = fmaxf(rmax[r], __shfl_xor(rmax[r], off, 32));
    }
    float p[8], rsum[8], scl[8];
#pragma unroll
    for (int r = 0; r < 8; ++r) {
      float mn = fmaxf(mrun[r], rmax[r]);
      scl[r] = __expf(mrun[r] - mn);
      p[r] = __expf(sv[r] - mn);
      rsum[r] = p[r];
      mrun[r] = mn;
    }
    for (int off = 1; off < 16; off <<= 1) {
#pragma unroll
      for (int r = 0; r < 8; ++r) rsum[r] += __shfl_xor(rsum[r], off, 32);
    }
#pragma unroll
    for (int r = 0; r < 8; ++r) {
      lrun[r] = lrun[r] * scl[r] + rsum[r];
      oacc[r] *= scl[r];
    }
    // stage P through LDS: C-layout (row=r+8*hi, col=lane) -> A-layout
#pragma unroll
    for (int r = 0; r < 8; ++r) psh[wid][(r + 8 * hi) * 16 + col] = (__bf16)p[r];
    v16bf pa = zero16bf();
#pragma unroll
    for (int v = 0; v < 8; ++v) {
      int jr = v * 2;
      pa[2 * v]     = psh[wid][col * 16 + jr];
      pa[2 * v + 1] = psh[wid][col * 16 + jr + 1];
    }
    // V fragment: B[jrel][d] = V[j0+jrel][d]; lanes hi==0 carry jrel=0..15
    v16bf vf = zero16bf();
    if (hi == 0) {
#pragma unroll
      for (int v = 0; v < 8; ++v) {
        int jg = j0 + 2 * v;
        vf[2 * v]     = (jg < cN)     ? Vb[(size_t)jg * cD + head * 16 + col]       : (__bf16)0.f;
        vf[2 * v + 1] = (jg + 1 < cN) ? Vb[(size_t)(jg + 1) * cD + head * 16 + col] : (__bf16)0.f;
      }
    }
    oacc = wmma_bf16(pa, vf, oacc);
  }
  int mb = m0 + hi * 8;
#pragma unroll
  for (int r = 0; r < 8; ++r) {
    int m = mb + r;
    if (m < cN)
      O[(size_t)b * cN * cD + (size_t)m * cD + head * 16 + col] = (__bf16)(oacc[r] / lrun[r]);
  }
}

// ============ stage 4: O projection + residual into h (f32 + bf16 shadow) =====
__global__ void k_oproj(const __bf16* __restrict__ ao, const __bf16* __restrict__ Wo16,
                        const float* __restrict__ bo, float* __restrict__ h,
                        __bf16* __restrict__ h16) {
  int b = blockIdx.y, lane = threadIdx.x & 31, wid = threadIdx.x >> 5;
  int m0 = blockIdx.x * 128 + wid * 16;
  v8f acc[8];
#pragma unroll
  for (int i = 0; i < 8; ++i) acc[i] = zero8();
  mm_strip(ao + (size_t)b * cN * cD, cD, m0, cN, Wo16, cD, 0, cD, lane, acc);
  int col = lane & 15, hi = lane >> 4, mb = m0 + hi * 8;
  float*  hb  = h   + (size_t)b * cN * cD;
  __bf16* hb16= h16 + (size_t)b * cN * cD;
#pragma unroll
  for (int nt = 0; nt < 8; ++nt) {
    int n = nt * 16 + col;
    float bv_ = bo[n];
#pragma unroll
    for (int r = 0; r < 8; ++r) {
      int m = mb + r;
      if (m < cN) {
        float val = hb[(size_t)m * cD + n] + acc[nt][r] + bv_;
        hb[(size_t)m * cD + n]   = val;
        hb16[(size_t)m * cD + n] = (__bf16)val;
      }
    }
  }
}

// ================= stage 5: FF1 (relu) -> ff bf16 [B,N,512] ===================
__global__ void k_ff1(const __bf16* __restrict__ h16, const __bf16* __restrict__ W116,
                      const float* __restrict__ b1, __bf16* __restrict__ ff) {
  int b = blockIdx.y, lane = threadIdx.x & 31, wid = threadIdx.x >> 5;
  int m0 = blockIdx.x * 128 + wid * 16;
  int n0g = blockIdx.z * 128;
  v8f acc[8];
#pragma unroll
  for (int i = 0; i < 8; ++i) acc[i] = zero8();
  mm_strip(h16 + (size_t)b * cN * cD, cD, m0, cN, W116, cD, n0g, cD, lane, acc);
  int col = lane & 15, hi = lane >> 4, mb = m0 + hi * 8;
  __bf16* fb = ff + (size_t)b * cN * cFF;
#pragma unroll
  for (int nt = 0; nt < 8; ++nt) {
    int n = n0g + nt * 16 + col;
    float bv_ = b1[n];
#pragma unroll
    for (int r = 0; r < 8; ++r) {
      int m = mb + r;
      if (m < cN) fb[(size_t)m * cFF + n] = (__bf16)fmaxf(acc[nt][r] + bv_, 0.0f);
    }
  }
}

// ============ stage 6: FF2 + residual -> final node_embed (bf16 shadow) =======
__global__ void k_ff2(const __bf16* __restrict__ ff, const __bf16* __restrict__ W216,
                      const float* __restrict__ b2, const float* __restrict__ h,
                      __bf16* __restrict__ h16) {
  int b = blockIdx.y, lane = threadIdx.x & 31, wid = threadIdx.x >> 5;
  int m0 = blockIdx.x * 128 + wid * 16;
  v8f acc[8];
#pragma unroll
  for (int i = 0; i < 8; ++i) acc[i] = zero8();
  mm_strip(ff + (size_t)b * cN * cFF, cFF, m0, cN, W216, cFF, 0, cFF, lane, acc);
  int col = lane & 15, hi = lane >> 4, mb = m0 + hi * 8;
  const float* hb = h + (size_t)b * cN * cD;
  __bf16* hb16 = h16 + (size_t)b * cN * cD;
#pragma unroll
  for (int nt = 0; nt < 8; ++nt) {
    int n = nt * 16 + col;
    float bv_ = b2[n];
#pragma unroll
    for (int r = 0; r < 8; ++r) {
      int m = mb + r;
      if (m < cN) {
        float val = hb[(size_t)m * cD + n] + acc[nt][r] + bv_;
        hb16[(size_t)m * cD + n] = (__bf16)val;   // only bf16 copy needed downstream
      }
    }
  }
}

// ============ stage 7: mem = memory + W_mem @ node_embed (transposed out) =====
__global__ void k_mem(const __bf16* __restrict__ h16, const float* __restrict__ memin,
                      const __bf16* __restrict__ Wm16, const float* __restrict__ bm,
                      float* __restrict__ memw) {
  int b = blockIdx.y, lane = threadIdx.x & 31, wid = threadIdx.x >> 5;
  int m0 = blockIdx.x * 128 + wid * 16;
  int n0g = blockIdx.z * 128;          // output-channel group (3D=384 -> 3 groups)
  v8f acc[8];
#pragma unroll
  for (int i = 0; i < 8; ++i) acc[i] = zero8();
  mm_strip(h16 + (size_t)b * cN * cD, cD, m0, cN, Wm16, cD, n0g, cD, lane, acc);
  int col = lane & 15, hi = lane >> 4, mb = m0 + hi * 8;
#pragma unroll
  for (int nt = 0; nt < 8; ++nt) {
    int o = n0g + nt * 16 + col;
    float bv_ = bm[o];
    size_t base = (size_t)b * 3 * cD * cN + (size_t)o * cN;
#pragma unroll
    for (int r = 0; r < 8; ++r) {
      int m = mb + r;
      if (m < cN) memw[base + m] = memin[base + m] + acc[nt][r] + bv_;
    }
  }
}

// ================= stage 8: context vector ====================================
__global__ void k_context(const float* __restrict__ depot, const float* __restrict__ graph,
                          const float* __restrict__ nef, const int* __restrict__ next_agent,
                          const int* __restrict__ position, const float* __restrict__ loadv,
                          const float* __restrict__ maxload, const float* __restrict__ speed,
                          const float* __restrict__ Wc, const float* __restrict__ bc,
                          float* __restrict__ ctx) {
  int b = blockIdx.x, d = threadIdx.x;
  int na = next_agent[b];
  int apos = position[b * cA + na];
  float s = bc[d] + depot[b * cD + d] + graph[b * cD + d];
  const float* wr = Wc + (size_t)d * (cD + 3);
  const float* nb = nef + (size_t)b * cD * cN;
  for (int j = 0; j < cD; ++j) s += wr[j] * nb[(size_t)j * cN + apos];
  s += wr[cD] * loadv[b * cA + na] + wr[cD + 1] * maxload[b * cA + na] +
       wr[cD + 2] * speed[b * cA + na];
  ctx[b * cD + d] = s;
}

// ================= stage 9: single-query MHA over glimpse key/val =============
__global__ void k_mha(const float* __restrict__ memw, const float* __restrict__ ctx,
                      float* __restrict__ mha) {
  int b = blockIdx.x;
  int lane = threadIdx.x & 31, head = threadIdx.x >> 5;  // 8 waves = 8 heads
  float qv[16];
#pragma unroll
  for (int d = 0; d < 16; ++d) qv[d] = ctx[b * cD + head * 16 + d];
  const float* gk = memw + (size_t)b * 3 * cD * cN + (size_t)head * 16 * cN;
  const float* gv = gk + (size_t)cD * cN;
  float m = -1e30f, l = 0.f, acc[16];
#pragma unroll
  for (int d = 0; d < 16; ++d) acc[d] = 0.f;
  for (int n = lane; n < cN; n += 32) {
    float s = 0.f;
#pragma unroll
    for (int d = 0; d < 16; ++d) s += gk[(size_t)d * cN + n] * qv[d];
    s *= 0.25f;                              // 1/sqrt(dk)
    float mn = fmaxf(m, s);
    float sc = __expf(m - mn), p = __expf(s - mn);
    l = l * sc + p;
#pragma unroll
    for (int d = 0; d < 16; ++d) acc[d] = acc[d] * sc + p * gv[(size_t)d * cN + n];
    m = mn;
  }
  for (int off = 16; off > 0; off >>= 1) {
    float mo = __shfl_xor(m, off, 32);
    float lo = __shfl_xor(l, off, 32);
    float mn = fmaxf(m, mo);
    float sa = __expf(m - mn), sb = __expf(mo - mn);
    l = l * sa + lo * sb;
#pragma unroll
    for (int d = 0; d < 16; ++d) {
      float ao = __shfl_xor(acc[d], off, 32);
      acc[d] = acc[d] * sa + ao * sb;
    }
    m = mn;
  }
  if (lane == 0) {
#pragma unroll
    for (int d = 0; d < 16; ++d) mha[b * cD + head * 16 + d] = acc[d] / l;
  }
}

// ================= stage 10: glimpse projection ===============================
__global__ void k_glimpse(const float* __restrict__ mha, const float* __restrict__ Wg,
                          const float* __restrict__ bg, float* __restrict__ gq) {
  int b = blockIdx.x, d = threadIdx.x;
  float s = bg[d];
  for (int j = 0; j < cD; ++j) s += Wg[(size_t)d * cD + j] * mha[b * cD + j];
  gq[b * cD + d] = s;
}

// ================= stage 11: logits + log_softmax =============================
__global__ void k_logits(const float* __restrict__ memw, const float* __restrict__ gq,
                         const float* __restrict__ mask, float* __restrict__ out) {
  __shared__ float red[256];
  __shared__ float gqs[128];
  int b = blockIdx.x, tid = threadIdx.x;
  if (tid < 128) gqs[tid] = gq[b * cD + tid];
  __syncthreads();
  const float* lk = memw + (size_t)b * 3 * cD * cN + (size_t)2 * cD * cN;
  float vals[4];
  float lmax = -3.0e38f;
#pragma unroll
  for (int i = 0; i < 4; ++i) {
    int n = i * 256 + tid;
    float t = -3.0e38f;
    if (n < cN) {
      float dot = 0.f;
      for (int d = 0; d < cD; ++d) dot += lk[(size_t)d * cN + n] * gqs[d];
      t = tanhf(dot * 0.08838834764831845f) * 10.0f + logf(mask[b * cN + n]);
    }
    vals[i] = t;
    lmax = fmaxf(lmax, t);
  }
  red[tid] = lmax; __syncthreads();
  for (int s = 128; s > 0; s >>= 1) {
    if (tid < s) red[tid] = fmaxf(red[tid], red[tid + s]);
    __syncthreads();
  }
  float gmax = red[0]; __syncthreads();
  float lsum = 0.f;
#pragma unroll
  for (int i = 0; i < 4; ++i) {
    int n = i * 256 + tid;
    if (n < cN) lsum += __expf(vals[i] - gmax);
  }
  red[tid] = lsum; __syncthreads();
  for (int s = 128; s > 0; s >>= 1) {
    if (tid < s) red[tid] += red[tid + s];
    __syncthreads();
  }
  float lse = logf(red[0]);
#pragma unroll
  for (int i = 0; i < 4; ++i) {
    int n = i * 256 + tid;
    if (n < cN) out[(size_t)b * cN + n] = vals[i] - gmax - lse;
  }
}

// ============================== host launcher =================================
extern "C" void kernel_launch(void* const* d_in, const int* in_sizes, int n_in,
                              void* d_out, int out_size, void* d_ws, size_t ws_size,
                              hipStream_t stream) {
  (void)in_sizes; (void)n_in; (void)out_size; (void)ws_size;
  const float* location = (const float*)d_in[0];
  const float* max_load = (const float*)d_in[1];
  const float* speed    = (const float*)d_in[2];
  const int*   next_ag  = (const int*)d_in[3];
  const int*   position = (const int*)d_in[4];
  const float* tta      = (const float*)d_in[5];
  const float* loadv    = (const float*)d_in[6];
  const float* demand   = (const float*)d_in[7];
  const float* mask     = (const float*)d_in[8];
  const float* nef      = (const float*)d_in[9];
  const float* depot    = (const float*)d_in[10];
  const float* graph    = (const float*)d_in[11];
  const float* memory   = (const float*)d_in[12];
  const float* W_na  = (const float*)d_in[13];
  const float* b_na  = (const float*)d_in[14];
  const float* Wq    = (const float*)d_in[15];
  const float* bq    = (const float*)d_in[16];
  const float* Wk    = (const float*)d_in[17];
  const float* bk    = (const float*)d_in[18];
  const float* Wv    = (const float*)d_in[19];
  const float* bv    = (const float*)d_in[20];
  const float* Wo    = (const float*)d_in[21];
  const float* bo    = (const float*)d_in[22];
  const float* W_ff1 = (const float*)d_in[23];
  const float* b_ff1 = (const float*)d_in[24];
  const float* W_ff2 = (const float*)d_in[25];
  const float* b_ff2 = (const float*)d_in[26];
  const float* W_mem = (const float*)d_in[27];
  const float* b_mem = (const float*)d_in[28];
  const float* W_ctx = (const float*)d_in[29];
  const float* b_ctx = (const float*)d_in[30];
  const float* W_gl  = (const float*)d_in[31];
  const float* b_gl  = (const float*)d_in[32];

  char* ws = (char*)d_ws;
  // layout (bytes):
  //   h    : [0, 131,072,000)                f32  [B,N,128]
  //   q    : [131,072,000, +65,536,000)      bf16
  //   k    : [196,608,000, +65,536,000)      bf16
  //   v    : [262,144,000, +65,536,000)      bf16
  //   ao   : [327,680,000, +65,536,000)      bf16
  //   ff   : [393,216,000, +262,144,000)     bf16 [B,N,512]
  //   h16  : [655,360,000, +65,536,000)      bf16 shadow of h
  //   memw : reuses [131,072,000, +393,216,000) after ff2 (q..ff dead)
  //   smalls + bf16 weights at 720,896,000
  float*  h    = (float*)ws;
  __bf16* q    = (__bf16*)(ws + 131072000ull);
  __bf16* kbf  = (__bf16*)(ws + 196608000ull);
  __bf16* vbf  = (__bf16*)(ws + 262144000ull);
  __bf16* ao   = (__bf16*)(ws + 327680000ull);
  __bf16* ff   = (__bf16*)(ws + 393216000ull);
  __bf16* h16  = (__bf16*)(ws + 655360000ull);
  float*  memw = (float*)(ws + 131072000ull);
  float*  ctx  = (float*)(ws + 720896000ull);
  float*  mha  = (float*)(ws + 721027072ull);
  float*  gq   = (float*)(ws + 721158144ull);
  __bf16* wq16 = (__bf16*)(ws + 721289216ull);   //  32,768 B
  __bf16* wk16 = (__bf16*)(ws + 721321984ull);   //  32,768 B
  __bf16* wv16 = (__bf16*)(ws + 721354752ull);   //  32,768 B
  __bf16* wo16 = (__bf16*)(ws + 721387520ull);   //  32,768 B
  __bf16* w116 = (__bf16*)(ws + 721420288ull);   // 131,072 B
  __bf16* w216 = (__bf16*)(ws + 721551360ull);   // 131,072 B
  __bf16* wm16 = (__bf16*)(ws + 721682432ull);   //  98,304 B
  float*  out  = (float*)d_out;

  // stage 0: one-shot weight conversion (tiny)
  k_cvtw<<<dim3(64),  256, 0, stream>>>(Wq,    wq16, cD * cD);
  k_cvtw<<<dim3(64),  256, 0, stream>>>(Wk,    wk16, cD * cD);
  k_cvtw<<<dim3(64),  256, 0, stream>>>(Wv,    wv16, cD * cD);
  k_cvtw<<<dim3(64),  256, 0, stream>>>(Wo,    wo16, cD * cD);
  k_cvtw<<<dim3(256), 256, 0, stream>>>(W_ff1, w116, cFF * cD);
  k_cvtw<<<dim3(256), 256, 0, stream>>>(W_ff2, w216, cD * cFF);
  k_cvtw<<<dim3(192), 256, 0, stream>>>(W_mem, wm16, 3 * cD * cD);

  k_prep   <<<dim3(125, cB),    128, 0, stream>>>(location, speed, next_ag, position, tta,
                                                  loadv, demand, W_na, b_na, h, h16);
  k_qkv    <<<dim3(8, cB, 3),   256, 0, stream>>>(h16, wq16, bq, wk16, bk, wv16, bv,
                                                  q, kbf, vbf);
  k_flash  <<<dim3(16, cH, cB), 128, 0, stream>>>(q, kbf, vbf, ao);
  k_oproj  <<<dim3(8, cB),      256, 0, stream>>>(ao, wo16, bo, h, h16);
  k_ff1    <<<dim3(8, cB, 4),   256, 0, stream>>>(h16, w116, b_ff1, ff);
  k_ff2    <<<dim3(8, cB),      256, 0, stream>>>(ff, w216, b_ff2, h, h16);
  k_mem    <<<dim3(8, cB, 3),   256, 0, stream>>>(h16, memory, wm16, b_mem, memw);
  k_context<<<dim3(cB),         128, 0, stream>>>(depot, graph, nef, next_ag, position,
                                                  loadv, max_load, speed, W_ctx, b_ctx, ctx);
  k_mha    <<<dim3(cB),         256, 0, stream>>>(memw, ctx, mha);
  k_glimpse<<<dim3(cB),         128, 0, stream>>>(mha, W_gl, b_gl, gq);
  k_logits <<<dim3(cB),         256, 0, stream>>>(memw, gq, mask, out);
}
